// SlatewiseGRU_16673063043320
// MI455X (gfx1250) — matrix operand order repaired and checked by reading, not compiled, not run on hardware
//
#include <hip/hip_runtime.h>
#include <hip/hip_bf16.h>

// ---------- problem constants ----------
#define Bsz 128
#define Ssz 64
#define Kst 10
#define Dd  256        // hidden dim
#define G3  768        // 3*D gate dim
#define NROW (Bsz*Ssz) // 8192 rows
#define WPER (G3*Dd)   // 196608 bf16 elements per packed weight matrix
#define TPB  256       // 8 waves per block
#define NTW  6         // N-tiles per wave (8 waves * 6 = 48 tiles = 768 cols)

typedef __attribute__((ext_vector_type(16))) __bf16 v16bf;
typedef __attribute__((ext_vector_type(8)))  __bf16 v8bf;
typedef __attribute__((ext_vector_type(8)))  float  v8f;

// explicit global-address-space pointer => guaranteed global_load_b128
typedef __attribute__((address_space(1))) const v16bf ga_v16bf;
typedef ga_v16bf* gv16p;

__device__ __forceinline__ unsigned short f2bf(float f) {
    unsigned u = __float_as_uint(f);
    u += 0x7FFFu + ((u >> 16) & 1u);   // round-to-nearest-even
    return (unsigned short)(u >> 16);
}

// ---------------------------------------------------------------------------
// Pack W_ih[:, :256] and W_hh (fp32, row-major [768 x stride]) into bf16
// B-fragments for v_wmma_f32_16x16x32_bf16:
//   dst[((gt*8 + kt)*32 + L)*16 + j] = W[g = gt*16 + L%16][d = kt*32 + (L/16)*16 + j]
// so each lane's 16 K-values are contiguous (one 32B load in the GEMM).
// ---------------------------------------------------------------------------
__global__ void pack_weights(const float* __restrict__ Wih,
                             const float* __restrict__ Whh,
                             unsigned short* __restrict__ dst) {
    int tid = blockIdx.x * blockDim.x + threadIdx.x;
    if (tid >= 2 * WPER) return;
    int m  = tid / WPER;          // 0 = W_ih (stride 512), 1 = W_hh (stride 256)
    int t  = tid % WPER;
    int j  = t & 15;
    int L  = (t >> 4) & 31;
    int kt = (t >> 9) & 7;
    int gt = t >> 12;             // 0..47
    int g  = gt * 16 + (L & 15);
    int d  = kt * 32 + (L >> 4) * 16 + j;
    float v = m ? Whh[(size_t)g * Dd + d] : Wih[(size_t)g * (2 * Dd) + d];
    dst[tid] = f2bf(v);
}

// ---------------------------------------------------------------------------
// Fused GRU recurrence. 512 blocks x 256 threads; each block owns 16 rows
// (one WMMA M-tile) for all K=10 steps. Per step, each of the 8 waves
// computes 6 of the 48 gate N-tiles with 16x16x32 bf16 WMMA, fp32 accum.
// B-fragments are batch-loaded per kt (6 x 32B) so loads pipeline ahead of
// the WMMAs; an opaque per-step offset stops LICM from hoisting+spilling
// the k-invariant weight stream.
// ---------------------------------------------------------------------------
__launch_bounds__(TPB)
__global__ void gru_kernel(const float* __restrict__ item_embs,   // [N,K,D]
                           const float* __restrict__ user_embs,   // [B,S,D]
                           const unsigned short* __restrict__ Wpk,// packed bf16
                           const float* __restrict__ b_ih,        // [768]
                           const float* __restrict__ b_hh,        // [768]
                           const float* __restrict__ w_out,       // [256]
                           const float* __restrict__ b_out,       // [1]
                           const int*   __restrict__ length,      // [B]
                           float* __restrict__ out) {             // [N,K]
    __shared__ alignas(16) unsigned short xbf[16][Dd + 8];  // bf16 x tile
    __shared__ alignas(16) unsigned short hbf[16][Dd + 8];  // bf16 h tile
    __shared__ alignas(16) float hf[16][Dd + 1];            // fp32 h
    __shared__ alignas(16) float gi[16][G3 + 4];            // x-gates
    __shared__ alignas(16) float gh[16][G3 + 4];            // h-gates
    __shared__ alignas(16) float bI[G3];                    // b_ih staged
    __shared__ alignas(16) float bH[G3];                    // b_hh staged
    __shared__ alignas(16) float wo[Dd];                    // w_out staged
    __shared__ float bo;                                    // b_out staged

    const int tid  = threadIdx.x;
    const int wave = tid >> 5;                 // 0..7
    const int L    = tid & 31;
    const int n0   = blockIdx.x * 16;          // first row of this tile
    const int b    = n0 / Ssz;                 // same batch elem for all 16 rows

    // ---- stage biases / readout weights in LDS (kills k-loop LICM bait) ----
    for (int c = tid; c < G3; c += TPB) { bI[c] = b_ih[c]; bH[c] = b_hh[c]; }
    for (int c = tid; c < Dd; c += TPB) wo[c] = w_out[c];
    if (tid == 0) bo = b_out[0];

    // ---- h0 = user_embs[b, clip(length[b]-1, 0)], broadcast to 16 rows ----
    int idx = length[b] - 1;
    if (idx < 0) idx = 0;
    const float* u0 = user_embs + ((size_t)b * Ssz + idx) * Dd;
    for (int c = tid; c < Dd; c += TPB) {
        float v = u0[c];
        unsigned short bv = f2bf(v);
        #pragma unroll
        for (int r = 0; r < 16; ++r) { hf[r][c] = v; hbf[r][c] = bv; }
    }
    __syncthreads();

    const unsigned short* wLane = Wpk + (size_t)L * 16;  // per-lane fragment base
    const int aRow = L & 15;
    const int aOff = (L >> 4) * 16;

    for (int k = 0; k < Kst; ++k) {
        // Opaque zero, fresh each step: weight-fragment addresses become
        // loop-variant so LICM cannot hoist+spill them, but the pointer
        // chain to the kernarg (global) is preserved.
        unsigned koff = 0;
        asm volatile("" : "+v"(koff));
        const unsigned short* wIk = wLane + koff;
        const unsigned short* wHk = wLane + koff + (size_t)WPER;

        // ---- load + convert x tile (only first D of the 2D input is nonzero) ----
        for (int e = tid; e < 16 * Dd; e += TPB) {
            int r = e >> 8, c = e & 255;
            size_t gidx = (((size_t)(n0 + r)) * Kst + k) * Dd + c;
            xbf[r][c] = f2bf(item_embs[gidx]);
            if (k + 1 < Kst) __builtin_prefetch(&item_embs[gidx + Dd], 0, 3);
        }
        __syncthreads();

        v8f acc[NTW] = {};

        // ---- gi = x @ W_ih[:, :D].T ----
        #pragma unroll
        for (int kt = 0; kt < 8; ++kt) {
            v8bf lo = *(const v8bf*)&xbf[aRow][kt * 32 + aOff];
            v8bf hi = *(const v8bf*)&xbf[aRow][kt * 32 + aOff + 8];
            v16bf A = __builtin_shufflevector(lo, hi,
                       0,1,2,3,4,5,6,7,8,9,10,11,12,13,14,15);
            v16bf Bfs[NTW];
            #pragma unroll
            for (int i = 0; i < NTW; ++i) {
                int nt = wave * NTW + i;
                Bfs[i] = *(gv16p)(unsigned long long)
                             (wIk + (size_t)(nt * 8 + kt) * 512);
            }
            #pragma unroll
            for (int i = 0; i < NTW; ++i)
                acc[i] = __builtin_amdgcn_wmma_f32_16x16x32_bf16(
                            false, A, false, Bfs[i], (short)0, acc[i],
                            false, false);
        }
        #pragma unroll
        for (int i = 0; i < NTW; ++i) {
            int col = (wave * NTW + i) * 16 + (L & 15);
            #pragma unroll
            for (int v = 0; v < 8; ++v) gi[(L >> 4) * 8 + v][col] = acc[i][v];
        }

        // ---- gh = h @ W_hh.T ----
        #pragma unroll
        for (int i = 0; i < NTW; ++i) acc[i] = v8f{};
        #pragma unroll
        for (int kt = 0; kt < 8; ++kt) {
            v8bf lo = *(const v8bf*)&hbf[aRow][kt * 32 + aOff];
            v8bf hi = *(const v8bf*)&hbf[aRow][kt * 32 + aOff + 8];
            v16bf A = __builtin_shufflevector(lo, hi,
                       0,1,2,3,4,5,6,7,8,9,10,11,12,13,14,15);
            v16bf Bfs[NTW];
            #pragma unroll
            for (int i = 0; i < NTW; ++i) {
                int nt = wave * NTW + i;
                Bfs[i] = *(gv16p)(unsigned long long)
                             (wHk + (size_t)(nt * 8 + kt) * 512);
            }
            #pragma unroll
            for (int i = 0; i < NTW; ++i)
                acc[i] = __builtin_amdgcn_wmma_f32_16x16x32_bf16(
                            false, A, false, Bfs[i], (short)0, acc[i],
                            false, false);
        }
        #pragma unroll
        for (int i = 0; i < NTW; ++i) {
            int col = (wave * NTW + i) * 16 + (L & 15);
            #pragma unroll
            for (int v = 0; v < 8; ++v) gh[(L >> 4) * 8 + v][col] = acc[i][v];
        }
        __syncthreads();

        // ---- fused gates: r,z = sigmoid; n = tanh(i_n + r*h_n); h update ----
        for (int e = tid; e < 16 * Dd; e += TPB) {
            int r = e >> 8, c = e & 255;
            float ir = gi[r][c]       + bI[c];
            float iz = gi[r][c + 256] + bI[c + 256];
            float in_= gi[r][c + 512] + bI[c + 512];
            float hr = gh[r][c]       + bH[c];
            float hz = gh[r][c + 256] + bH[c + 256];
            float hn = gh[r][c + 512] + bH[c + 512];
            float rg = 1.f / (1.f + __expf(-(ir + hr)));
            float zg = 1.f / (1.f + __expf(-(iz + hz)));
            float a  = in_ + rg * hn;
            float e2 = __expf(-2.f * a);
            float nn = (1.f - e2) / (1.f + e2);
            float hnew = (1.f - zg) * nn + zg * hf[r][c];
            hf[r][c]  = hnew;
            hbf[r][c] = f2bf(hnew);
        }
        __syncthreads();

        // ---- y[n] = h . w_out + b_out  (16 lanes per row, shfl reduce) ----
        {
            int r   = tid >> 4;   // 0..15
            int g16 = tid & 15;
            float p = 0.f;
            for (int c = g16; c < Dd; c += 16) p += hf[r][c] * wo[c];
            p += __shfl_xor(p, 1, 32);
            p += __shfl_xor(p, 2, 32);
            p += __shfl_xor(p, 4, 32);
            p += __shfl_xor(p, 8, 32);
            if (g16 == 0) out[(size_t)(n0 + r) * Kst + k] = p + bo;
        }
        __syncthreads();
    }
}

extern "C" void kernel_launch(void* const* d_in, const int* in_sizes, int n_in,
                              void* d_out, int out_size, void* d_ws, size_t ws_size,
                              hipStream_t stream) {
    (void)in_sizes; (void)n_in; (void)out_size; (void)ws_size;
    const float* item = (const float*)d_in[0];
    const float* user = (const float*)d_in[1];
    const float* Wih  = (const float*)d_in[2];
    const float* Whh  = (const float*)d_in[3];
    const float* bih  = (const float*)d_in[4];
    const float* bhh  = (const float*)d_in[5];
    const float* wout = (const float*)d_in[6];
    const float* bout = (const float*)d_in[7];
    const int*   len  = (const int*)d_in[8];
    unsigned short* wpk = (unsigned short*)d_ws;   // 2*196608*2 = 768 KB

    pack_weights<<<(2 * WPER + 255) / 256, 256, 0, stream>>>(Wih, Whh, wpk);
    gru_kernel<<<NROW / 16, TPB, 0, stream>>>(item, user, wpk, bih, bhh,
                                              wout, bout, len, (float*)d_out);
}
